// BusStopPredictor_9431748182756
// MI455X (gfx1250) — compile-verified, matrix operand order checked by reading, not run on hardware
//
#include <hip/hip_runtime.h>
#include <hip/hip_bf16.h>
#include <stdint.h>

#define N_NODES 1000000
#define N_EDGES 2500000
#define EPS_BN 1e-5f

typedef __bf16 bf16_t;
typedef __attribute__((ext_vector_type(16))) bf16_t v16bf;
typedef __attribute__((ext_vector_type(8)))  bf16_t v8bf;
typedef __attribute__((ext_vector_type(8)))  float  v8f;

union Frag { v16bf v; v8bf h[2]; };

__device__ __forceinline__ uint16_t f2bf_bits(float f) {
    uint32_t u = __builtin_bit_cast(uint32_t, f);
    uint32_t r = u + 0x7FFFu + ((u >> 16) & 1u);   // round-to-nearest-even
    return (uint16_t)(r >> 16);
}

// ---------------- workspace zeroing ----------------
__global__ void zero_kernel(float* __restrict__ p, size_t n) {
    size_t i = (size_t)blockIdx.x * blockDim.x + threadIdx.x;
    size_t stride = (size_t)gridDim.x * blockDim.x;
    for (; i < n; i += stride) p[i] = 0.0f;
}

// ---------------- weight f32 -> bf16 ----------------
__global__ void wconv_kernel(const float* __restrict__ Wl, const float* __restrict__ Wr,
                             bf16_t* __restrict__ Wlb, bf16_t* __restrict__ Wrb) {
    int i = blockIdx.x * blockDim.x + threadIdx.x;
    if (i < 4096)      Wlb[i] = (bf16_t)Wl[i];
    else if (i < 8192) Wrb[i - 4096] = (bf16_t)Wr[i - 4096];
}

// ---------------- fused: degree count + layer-1 scatter (single edge pass) ----------------
__global__ void edge_pass1_kernel(const long long* __restrict__ ei,
                                  const float* __restrict__ x,
                                  float* __restrict__ deg, float* __restrict__ agg1) {
    long long e = (long long)blockIdx.x * blockDim.x + threadIdx.x;
    if (e >= N_EDGES) return;
    size_t s = (size_t)ei[e];
    size_t d = (size_t)ei[N_EDGES + e];
    float4 v = *(const float4*)(x + s * 4);
    atomicAdd(&deg[d], 1.0f);
    atomicAdd(&agg1[d * 4 + 0], v.x);
    atomicAdd(&agg1[d * 4 + 1], v.y);
    atomicAdd(&agg1[d * 4 + 2], v.z);
    atomicAdd(&agg1[d * 4 + 3], v.w);
}

__global__ void invdeg_kernel(float* __restrict__ deg) {
    size_t i = (size_t)blockIdx.x * blockDim.x + threadIdx.x;
    if (i >= N_NODES) return;
    deg[i] = 1.0f / fmaxf(deg[i], 1.0f);
}

// ---------------- layer-1 fused: lin_l + lin_r + BN + ReLU -> h1 (bf16) ----------------
__global__ void layer1_kernel(const float* __restrict__ x, const float* __restrict__ agg1,
                              const float* __restrict__ inv_deg,
                              const float* __restrict__ W1l, const float* __restrict__ b1,
                              const float* __restrict__ W1r,
                              const float* __restrict__ g1, const float* __restrict__ be1,
                              const float* __restrict__ rm1, const float* __restrict__ rv1,
                              bf16_t* __restrict__ h1) {
    size_t n = (size_t)blockIdx.x * blockDim.x + threadIdx.x;
    if (n >= N_NODES) return;
    float4 xr = *(const float4*)(x + n * 4);
    float  id = inv_deg[n];
    float4 a  = *(const float4*)(agg1 + n * 4);
    a.x *= id; a.y *= id; a.z *= id; a.w *= id;

    uint32_t packed[32];
#pragma unroll
    for (int o = 0; o < 64; ++o) {
        float4 wl = *(const float4*)(W1l + o * 4);
        float4 wr = *(const float4*)(W1r + o * 4);
        float v = b1[o]
                + wl.x * a.x + wl.y * a.y + wl.z * a.z + wl.w * a.w
                + wr.x * xr.x + wr.y * xr.y + wr.z * xr.z + wr.w * xr.w;
        float sc = g1[o] * rsqrtf(rv1[o] + EPS_BN);
        v = (v - rm1[o]) * sc + be1[o];
        v = fmaxf(v, 0.0f);
        uint16_t b = f2bf_bits(v);
        if (o & 1) packed[o >> 1] |= ((uint32_t)b) << 16;
        else       packed[o >> 1] = (uint32_t)b;
    }
    uint4* dst = (uint4*)(h1 + n * 64);
#pragma unroll
    for (int i = 0; i < 8; ++i) dst[i] = ((const uint4*)packed)[i];
}

// ---------------- layer-2 scatter: h1 (bf16, L2-resident) -> agg2 (f32 atomics) ----------------
__global__ void scatter_h1_kernel(const long long* __restrict__ ei,
                                  const bf16_t* __restrict__ h1, float* __restrict__ agg2) {
    long long t = (long long)blockIdx.x * blockDim.x + threadIdx.x;
    long long e = t >> 3;
    int part = (int)(t & 7);
    if (e >= N_EDGES) return;
    size_t s = (size_t)ei[e];
    size_t d = (size_t)ei[N_EDGES + e];
    v8bf h = *(const v8bf*)(h1 + s * 64 + part * 8);
    float* ap = agg2 + d * 64 + part * 8;
#pragma unroll
    for (int i = 0; i < 8; ++i) atomicAdd(ap + i, (float)h[i]);
}

// ---------------- layer-2 fused WMMA ----------------
// Weights staged in LDS (16 KB per block); one wave per 64-node super-tile (4 M-tiles).
// Per tile: 4 accumulating v_wmma_f32_16x16x32_bf16 chains over K=64 for lin_l and lin_r,
// B fragments pulled from LDS via ds_load_b128 (keeps VGPR pressure low, no spills).
// D = (agg2*inv_deg)@W2l.T + h1@W2r.T + b2 -> BN -> ReLU -> @Wp.T + bp -> sigmoid.
__global__ void layer2_wmma_kernel(const float* __restrict__ agg2,
                                   const bf16_t* __restrict__ h1,
                                   const float* __restrict__ inv_deg,
                                   const bf16_t* __restrict__ Wl,
                                   const bf16_t* __restrict__ Wr,
                                   const float* __restrict__ b2,
                                   const float* __restrict__ g2,
                                   const float* __restrict__ be2,
                                   const float* __restrict__ rm2,
                                   const float* __restrict__ rv2,
                                   const float* __restrict__ Wp,
                                   const float* __restrict__ bp,
                                   float* __restrict__ out) {
    __shared__ bf16_t lds_w[8192];             // [0..4095] = W2l.bf16, [4096..8191] = W2r.bf16

    // Cooperative staging: 16 KB as 1024 x uint4, 4 per thread.
    {
        const uint4* sl = (const uint4*)Wl;    // 512 uint4
        const uint4* sr = (const uint4*)Wr;    // 512 uint4
        uint4* dst = (uint4*)lds_w;
        for (int i = threadIdx.x; i < 1024; i += 256)
            dst[i] = (i < 512) ? sl[i] : sr[i - 512];
    }
    __syncthreads();

    const int wave = (blockIdx.x * blockDim.x + threadIdx.x) >> 5;
    const int lane = threadIdx.x & 31;
    const int n_super = N_NODES / 64;
    if (wave >= n_super) return;               // wave-uniform: EXEC stays all-1s below

    const int row   = lane & 15;               // A-matrix M index / B-matrix N index
    const int khalf = lane >> 4;               // which 8-wide K subgroup this lane holds
    const size_t base = (size_t)wave * 64;
    const bf16_t* wl_lds = lds_w;
    const bf16_t* wr_lds = lds_w + 4096;

    // Per-column epilogue params (per-lane, reused by all 4 tiles).
    float biasv[4], scv[4], shv[4], wpv[4];
#pragma unroll
    for (int nt = 0; nt < 4; ++nt) {
        const int col = nt * 16 + row;
        biasv[nt] = b2[col];
        scv[nt]   = g2[col] * rsqrtf(rv2[col] + EPS_BN);
        shv[nt]   = be2[col] - rm2[col] * scv[nt];
        wpv[nt]   = Wp[col];
    }
    const float bpv = bp[0];

#pragma unroll
    for (int t = 0; t < 4; ++t) {
        const size_t node = base + t * 16 + row;
        if (t < 3) {   // prefetch next tile's A rows (global_prefetch_b8)
            __builtin_prefetch(agg2 + (node + 16) * 64, 0, 3);
            __builtin_prefetch(h1 + (node + 16) * 64, 0, 3);
        }
        const float id = inv_deg[node];

        // ---- Build A fragments (ISA 16-bit A layout: lane holds row = lane%16,
        //      K subgroups selected by lane/16). ----
        Frag Aagg[2], Ah[2];
#pragma unroll
        for (int kc = 0; kc < 2; ++kc) {
            const int kb0 = kc * 32 + khalf * 8;        // elems 0..7  (VGPR 0-3)
            const int kb1 = kc * 32 + 16 + khalf * 8;   // elems 8..15 (VGPR 4-7)
            Ah[kc].h[0] = *(const v8bf*)(h1 + node * 64 + kb0);
            Ah[kc].h[1] = *(const v8bf*)(h1 + node * 64 + kb1);

            const float4 q0 = *(const float4*)(agg2 + node * 64 + kb0);
            const float4 q1 = *(const float4*)(agg2 + node * 64 + kb0 + 4);
            const float4 q2 = *(const float4*)(agg2 + node * 64 + kb1);
            const float4 q3 = *(const float4*)(agg2 + node * 64 + kb1 + 4);
            v16bf a;
            a[0]  = (bf16_t)(q0.x * id); a[1]  = (bf16_t)(q0.y * id);
            a[2]  = (bf16_t)(q0.z * id); a[3]  = (bf16_t)(q0.w * id);
            a[4]  = (bf16_t)(q1.x * id); a[5]  = (bf16_t)(q1.y * id);
            a[6]  = (bf16_t)(q1.z * id); a[7]  = (bf16_t)(q1.w * id);
            a[8]  = (bf16_t)(q2.x * id); a[9]  = (bf16_t)(q2.y * id);
            a[10] = (bf16_t)(q2.z * id); a[11] = (bf16_t)(q2.w * id);
            a[12] = (bf16_t)(q3.x * id); a[13] = (bf16_t)(q3.y * id);
            a[14] = (bf16_t)(q3.z * id); a[15] = (bf16_t)(q3.w * id);
            Aagg[kc].v = a;
        }

        float proj[8];
#pragma unroll
        for (int j = 0; j < 8; ++j) proj[j] = 0.0f;

#pragma unroll
        for (int nt = 0; nt < 4; ++nt) {
            const int col = nt * 16 + row;
            // B = W.T : B[k][col] = W[col*64 + k]; 16-byte chunks from LDS (ds_load_b128).
            Frag Bl0, Bl1, Br0, Br1;
            Bl0.h[0] = *(const v8bf*)(wl_lds + col * 64 +      khalf * 8);
            Bl0.h[1] = *(const v8bf*)(wl_lds + col * 64 + 16 + khalf * 8);
            Bl1.h[0] = *(const v8bf*)(wl_lds + col * 64 + 32 + khalf * 8);
            Bl1.h[1] = *(const v8bf*)(wl_lds + col * 64 + 48 + khalf * 8);
            Br0.h[0] = *(const v8bf*)(wr_lds + col * 64 +      khalf * 8);
            Br0.h[1] = *(const v8bf*)(wr_lds + col * 64 + 16 + khalf * 8);
            Br1.h[0] = *(const v8bf*)(wr_lds + col * 64 + 32 + khalf * 8);
            Br1.h[1] = *(const v8bf*)(wr_lds + col * 64 + 48 + khalf * 8);

            v8f c = {};
            c = __builtin_amdgcn_wmma_f32_16x16x32_bf16(false, Aagg[0].v, false, Bl0.v, (short)0, c, false, false);
            c = __builtin_amdgcn_wmma_f32_16x16x32_bf16(false, Aagg[1].v, false, Bl1.v, (short)0, c, false, false);
            c = __builtin_amdgcn_wmma_f32_16x16x32_bf16(false, Ah[0].v,   false, Br0.v, (short)0, c, false, false);
            c = __builtin_amdgcn_wmma_f32_16x16x32_bf16(false, Ah[1].v,   false, Br1.v, (short)0, c, false, false);

#pragma unroll
            for (int j = 0; j < 8; ++j) {      // C/D layout: VGPR j -> M = j + 8*khalf
                float v = (c[j] + biasv[nt]) * scv[nt] + shv[nt];
                v = fmaxf(v, 0.0f);
                proj[j] += v * wpv[nt];
            }
        }

        // Sum the 16 columns held across lanes of each half (offsets 1..8 stay in-half).
#pragma unroll
        for (int off = 1; off < 16; off <<= 1) {
#pragma unroll
            for (int j = 0; j < 8; ++j) proj[j] += __shfl_xor(proj[j], off, 32);
        }

        if (row == 0) {
#pragma unroll
            for (int j = 0; j < 8; ++j) {
                float z = proj[j] + bpv;
                out[base + t * 16 + khalf * 8 + j] = 1.0f / (1.0f + __expf(-z));
            }
        }
    }
}

extern "C" void kernel_launch(void* const* d_in, const int* in_sizes, int n_in,
                              void* d_out, int out_size, void* d_ws, size_t ws_size,
                              hipStream_t stream) {
    const float*     x   = (const float*)d_in[0];
    const long long* ei  = (const long long*)d_in[1];
    const float* W1l = (const float*)d_in[2];
    const float* b1  = (const float*)d_in[3];
    const float* W1r = (const float*)d_in[4];
    const float* g1  = (const float*)d_in[5];
    const float* be1 = (const float*)d_in[6];
    const float* rm1 = (const float*)d_in[7];
    const float* rv1 = (const float*)d_in[8];
    const float* W2l = (const float*)d_in[9];
    const float* b2  = (const float*)d_in[10];
    const float* W2r = (const float*)d_in[11];
    const float* g2  = (const float*)d_in[12];
    const float* be2 = (const float*)d_in[13];
    const float* rm2 = (const float*)d_in[14];
    const float* rv2 = (const float*)d_in[15];
    const float* Wp  = (const float*)d_in[16];
    const float* bp  = (const float*)d_in[17];
    float* out = (float*)d_out;

    // Workspace layout: [deg/inv_deg N f32][agg1 4N f32][agg2 64N f32]
    //                   [h1 64N bf16][Wl 4096 bf16][Wr 4096 bf16]
    float*  ws_f    = (float*)d_ws;
    float*  inv_deg = ws_f;
    float*  agg1    = ws_f + (size_t)N_NODES;
    float*  agg2    = ws_f + (size_t)5 * N_NODES;
    bf16_t* h1      = (bf16_t*)(ws_f + (size_t)69 * N_NODES);
    bf16_t* Wlb     = h1 + (size_t)64 * N_NODES;
    bf16_t* Wrb     = Wlb + 4096;

    zero_kernel<<<32768, 256, 0, stream>>>(ws_f, (size_t)69 * N_NODES);
    wconv_kernel<<<32, 256, 0, stream>>>(W2l, W2r, Wlb, Wrb);
    edge_pass1_kernel<<<(N_EDGES + 255) / 256, 256, 0, stream>>>(ei, x, inv_deg, agg1);
    invdeg_kernel<<<(N_NODES + 255) / 256, 256, 0, stream>>>(inv_deg);
    layer1_kernel<<<(N_NODES + 255) / 256, 256, 0, stream>>>(
        x, agg1, inv_deg, W1l, b1, W1r, g1, be1, rm1, rv1, h1);
    scatter_h1_kernel<<<(N_EDGES * 8 + 255) / 256, 256, 0, stream>>>(ei, h1, agg2);

    const int n_super = N_NODES / 64;
    layer2_wmma_kernel<<<(n_super * 32 + 255) / 256, 256, 0, stream>>>(
        agg2, h1, inv_deg, Wlb, Wrb, b2, g2, be2, rm2, rv2, Wp, bp, out);
}
